// attention_1675037246000
// MI455X (gfx1250) — matrix operand order repaired and checked by reading, not compiled, not run on hardware
//
#include <hip/hip_runtime.h>
#include <hip/hip_bf16.h>

// ---------------------------------------------------------------------------
// Multi-head attention forward, CDNA5 (gfx1250), wave32 WMMA f16 w/ f32 accum.
//   x:[2,4096,768] f32, w_qkv:[2304,768] f32, w_proj:[768,768] f32
//   out:[2,4096,768] f32
// Flash-attention structure (scores never touch HBM). Attention K/V tiles are
// double-buffered in LDS, filled with gfx1250 async global->LDS loads
// (global_load_async_to_lds_b128, ASYNCcnt / s_wait_asynccnt), shared by all
// 8 waves of a 256-thread block => ~8x less L2 traffic than per-wave loads.
// Workspace: ~64.5 MiB of f16 staging (xh, wqh, wph, Q, K, V^T, attn-out).
// ---------------------------------------------------------------------------

typedef _Float16 half_t;
typedef __attribute__((ext_vector_type(16))) _Float16 v16h;
typedef __attribute__((ext_vector_type(8)))  float    v8f;

#define EMBED   768
#define QKV_OUT 2304
#define NSEQ    4096
#define NHEAD   8
#define HD      96
#define BATCH   2
#define SCALE_F 0.1020620726159658f   // 96^-0.5

union Frag16 { v16h h; unsigned u[8]; };

// A-fragment (16x32 MxK, f16), row-major source with leading dim ldm (halfs).
// ISA layout: lane L -> row m=L&15, kHalf=L>>4; VGPR v holds f16 pair at
// k = (v&3)*2 + kHalf*8 + (v>>2)*16 (pairs contiguous in K -> dword loads).
__device__ inline v16h load_a_frag(const half_t* src, int ldm) {
    const int lane = threadIdx.x & 31;
    const int m = lane & 15, hk = lane >> 4;
    const half_t* row = src + m * ldm;
    Frag16 f;
#pragma unroll
    for (int v = 0; v < 8; ++v) {
        const int k = ((v & 3) << 1) + (hk << 3) + ((v >> 2) << 4);
        f.u[v] = *(const unsigned*)(row + k);
    }
    return f.h;
}

// B-fragment (32x16 KxN, f16) where B[k][n] = src[n*ldn + k] (source is
// N-major with K contiguous — matches W[out,in] rows, K rows, V^T rows).
// ISA layout: lane L -> col n=L&15, kHalf=L>>4; VGPR v holds pair at
// k = (v&3)*2 + kHalf*8 + (v>>2)*16 (pairs contiguous in K -> dword loads).
__device__ inline v16h load_b_frag(const half_t* src, int ldn) {
    const int lane = threadIdx.x & 31;
    const int n = lane & 15, hk = lane >> 4;
    const half_t* row = src + n * ldn;
    Frag16 f;
#pragma unroll
    for (int v = 0; v < 8; ++v) {
        const int k = ((v & 3) << 1) + (hk << 3) + ((v >> 2) << 4);
        f.u[v] = *(const unsigned*)(row + k);
    }
    return f.h;
}

__device__ inline v8f wmma_f16(v16h a, v16h b, v8f c) {
    return __builtin_amdgcn_wmma_f32_16x16x32_f16(false, a, false, b,
                                                  (short)0, c, false, false);
}

// gfx1250 async global->LDS copy, 16B per lane, tracked with ASYNCcnt.
// VDST carries the LDS byte offset = low 32 bits of the generic LDS address
// (flat->LDS mapping truncates, ISA 10.2); VADDR is the 64-bit global address.
__device__ inline void async_copy_b128(void* lds, const void* gsrc) {
    unsigned loff = (unsigned)(size_t)lds;
    asm volatile("global_load_async_to_lds_b128 %0, %1, off"
                 :: "v"(loff), "v"(gsrc) : "memory");
}

// ---------------------------------------------------------------------------
__global__ void cvt_f32_to_f16(const float* __restrict__ src,
                               half_t* __restrict__ dst, int n) {
    int i = blockIdx.x * blockDim.x + threadIdx.x;
    if (i < n) dst[i] = (half_t)src[i];
}

// ---------------------------------------------------------------------------
// QKV GEMM: [8192x768] @ [2304x768]^T. 4 waves/block, wave = 16 rows x 64 cols.
// Epilogue scatters into Q[b,h,n,d], K[b,h,n,d], VT[b,h,d,n] (f16).
__global__ __launch_bounds__(128)
void qkv_gemm_kernel(const half_t* __restrict__ xh, const half_t* __restrict__ wqh,
                     half_t* __restrict__ qh, half_t* __restrict__ kh,
                     half_t* __restrict__ vth) {
    const int wave = threadIdx.x >> 5;
    const int lane = threadIdx.x & 31;
    const int rowBase = blockIdx.x * 64 + wave * 16;
    const int colBase = blockIdx.y * 64;

    v8f acc[4] = {};
    const half_t* aRow = xh + rowBase * EMBED;
    for (int k = 0; k < EMBED; k += 32) {
        v16h a = load_a_frag(aRow + k, EMBED);
#pragma unroll
        for (int t = 0; t < 4; ++t) {
            v16h b = load_b_frag(wqh + (colBase + t * 16) * EMBED + k, EMBED);
            acc[t] = wmma_f16(a, b, acc[t]);
        }
    }

    const int m0   = rowBase + 8 * (lane >> 4);
    const int ncol = lane & 15;
#pragma unroll
    for (int t = 0; t < 4; ++t) {
        const int o     = colBase + t * 16 + ncol;
        const int which = o / EMBED;            // 0=Q 1=K 2=V
        const int rem   = o % EMBED;
        const int h = rem / HD, d = rem % HD;
#pragma unroll
        for (int v = 0; v < 8; ++v) {
            const int m = m0 + v;
            const int b = m >> 12, n = m & (NSEQ - 1);
            const long bh = (long)b * NHEAD + h;
            const half_t val = (half_t)acc[t][v];
            if (which == 0)      qh[(bh * NSEQ + n) * HD + d] = val;
            else if (which == 1) kh[(bh * NSEQ + n) * HD + d] = val;
            else                 vth[(bh * HD + d) * NSEQ + n] = val;
        }
    }
}

// ---------------------------------------------------------------------------
// Flash attention: 256 threads = 8 waves, each wave owns 16 query rows of one
// (b,h); block covers 128 rows. K/V tiles (64 keys) double-buffered in LDS via
// async loads, shared by all waves. Online softmax; O accumulated in f32.
__global__ __launch_bounds__(256)
void flash_attn_kernel(const half_t* __restrict__ qh, const half_t* __restrict__ kh,
                       const half_t* __restrict__ vth, half_t* __restrict__ ah) {
    __shared__ __align__(16) half_t kt_s[2][64 * 96];   // K tile  [key][d]
    __shared__ __align__(16) half_t vt_s[2][96 * 64];   // V^T tile [d][key]
    __shared__ __align__(16) half_t p_s[8][16 * 32];    // per-wave P staging

    const int tid  = threadIdx.x;
    const int wave = tid >> 5;
    const int lane = tid & 31;
    const int bh    = blockIdx.x;                 // 0..15 (b*8+h)
    const int qbase = blockIdx.y * 128 + wave * 16;

    const half_t* Q  = qh  + (long)bh * NSEQ * HD;
    const half_t* K  = kh  + (long)bh * NSEQ * HD;
    const half_t* VT = vth + (long)bh * HD * NSEQ;

    // Q fragments for d = 0..95 (3 x K32), loaded once per wave.
    v16h aq[3];
#pragma unroll
    for (int c = 0; c < 3; ++c)
        aq[c] = load_a_frag(Q + (long)qbase * HD + c * 32, HD);

    float mrow[8], lrow[8];
    v8f o[6] = {};
#pragma unroll
    for (int v = 0; v < 8; ++v) { mrow[v] = -1e30f; lrow[v] = 0.0f; }

    half_t* P = p_s[wave];
    const int mloc = 8 * (lane >> 4);
    const int nloc = lane & 15;

    // Cooperative async tile fill: 64x96 K tile (contiguous 12KB) and 96x64
    // V^T tile (96 rows of 128B), 768 b128 chunks each = 3+3 per thread.
    auto fill_tiles = [&](half_t* kt, half_t* vt, int kb) {
#pragma unroll
        for (int j = 0; j < 3; ++j) {
            const int c = tid + j * 256;
            async_copy_b128(kt + c * 8, K + (long)kb * HD + c * 8);
        }
#pragma unroll
        for (int j = 0; j < 3; ++j) {
            const int c = tid + j * 256;
            const int d = c >> 3, seg = c & 7;
            async_copy_b128(vt + c * 8, VT + (long)d * NSEQ + kb + seg * 8);
        }
    };

    fill_tiles(kt_s[0], vt_s[0], 0);
    int buf = 0;
    for (int kb = 0; kb < NSEQ; kb += 64, buf ^= 1) {
        const bool more = (kb + 64) < NSEQ;
        if (more) fill_tiles(kt_s[buf ^ 1], vt_s[buf ^ 1], kb + 64);
        // Retire the fill for the tile we are about to consume; keep the
        // prefetch for the next tile (6 async ops per thread) in flight.
        if (more) asm volatile("s_wait_asynccnt 0x6" ::: "memory");
        else      asm volatile("s_wait_asynccnt 0x0" ::: "memory");
        __syncthreads();

        const half_t* kt = kt_s[buf];
        const half_t* vt = vt_s[buf];
#pragma unroll
        for (int sub = 0; sub < 2; ++sub) {
            // ---- S = (Q K^T) * scale, two 16-col tiles (32 keys) ----------
            v8f s[2] = {};
#pragma unroll
            for (int t = 0; t < 2; ++t) {
                const half_t* krow = kt + (sub * 32 + t * 16) * HD;
#pragma unroll
                for (int c = 0; c < 3; ++c)
                    s[t] = wmma_f16(aq[c], load_b_frag(krow + c * 32, HD), s[t]);
            }
            // ---- online softmax (rows across 16 lanes of each half) -------
#pragma unroll
            for (int v = 0; v < 8; ++v) {
                float mx = fmaxf(s[0][v] * SCALE_F, s[1][v] * SCALE_F);
                mx = fmaxf(mx, __shfl_xor(mx, 1, 32));
                mx = fmaxf(mx, __shfl_xor(mx, 2, 32));
                mx = fmaxf(mx, __shfl_xor(mx, 4, 32));
                mx = fmaxf(mx, __shfl_xor(mx, 8, 32));
                const float mnew = fmaxf(mrow[v], mx);
                const float corr = __expf(mrow[v] - mnew);
                const float p0 = __expf(s[0][v] * SCALE_F - mnew);
                const float p1 = __expf(s[1][v] * SCALE_F - mnew);
                float rs = p0 + p1;
                rs += __shfl_xor(rs, 1, 32);
                rs += __shfl_xor(rs, 2, 32);
                rs += __shfl_xor(rs, 4, 32);
                rs += __shfl_xor(rs, 8, 32);
                lrow[v] = lrow[v] * corr + rs;
                mrow[v] = mnew;
#pragma unroll
                for (int t = 0; t < 6; ++t) o[t][v] *= corr;
                s[0][v] = p0; s[1][v] = p1;
            }
            // ---- P (C-layout) -> per-wave LDS -> A-fragment ---------------
            // Same-wave DS ops are in-order (ISA 4.2/7.3): no barrier needed.
#pragma unroll
            for (int v = 0; v < 8; ++v) {
                P[(mloc + v) * 32 + nloc]      = (half_t)s[0][v];
                P[(mloc + v) * 32 + 16 + nloc] = (half_t)s[1][v];
            }
            v16h ap = load_a_frag(P, 32);
            // ---- O += P V (B-frags from LDS V^T: K contiguous) ------------
#pragma unroll
            for (int t = 0; t < 6; ++t) {
                v16h bv = load_b_frag(vt + (t * 16) * 64 + sub * 32, 64);
                o[t] = wmma_f16(ap, bv, o[t]);
            }
        }
        __syncthreads();   // all waves done reading kt/vt before refill
    }

    // ---- normalize + store to attn-out [B,N,C] (f16 for proj GEMM) --------
    const int b = bh >> 3, h = bh & 7;
#pragma unroll
    for (int t = 0; t < 6; ++t) {
        const int d = t * 16 + nloc;
#pragma unroll
        for (int v = 0; v < 8; ++v) {
            const int n = qbase + mloc + v;
            ah[((long)b * NSEQ + n) * EMBED + h * HD + d] =
                (half_t)(o[t][v] / lrow[v]);
        }
    }
}

// ---------------------------------------------------------------------------
// Projection GEMM: out[8192x768] f32 = ah @ w_proj^T
__global__ __launch_bounds__(128)
void proj_gemm_kernel(const half_t* __restrict__ ah, const half_t* __restrict__ wph,
                      float* __restrict__ out) {
    const int wave = threadIdx.x >> 5;
    const int lane = threadIdx.x & 31;
    const int rowBase = blockIdx.x * 64 + wave * 16;
    const int colBase = blockIdx.y * 64;

    v8f acc[4] = {};
    const half_t* aRow = ah + (long)rowBase * EMBED;
    for (int k = 0; k < EMBED; k += 32) {
        v16h a = load_a_frag(aRow + k, EMBED);
#pragma unroll
        for (int t = 0; t < 4; ++t) {
            v16h b = load_b_frag(wph + (colBase + t * 16) * EMBED + k, EMBED);
            acc[t] = wmma_f16(a, b, acc[t]);
        }
    }
    const int m0   = rowBase + 8 * (lane >> 4);
    const int ncol = lane & 15;
#pragma unroll
    for (int t = 0; t < 4; ++t)
#pragma unroll
        for (int v = 0; v < 8; ++v)
            out[(long)(m0 + v) * EMBED + colBase + t * 16 + ncol] = acc[t][v];
}

// ---------------------------------------------------------------------------
extern "C" void kernel_launch(void* const* d_in, const int* in_sizes, int n_in,
                              void* d_out, int out_size, void* d_ws, size_t ws_size,
                              hipStream_t stream) {
    const float* x      = (const float*)d_in[0];   // [2,4096,768]
    const float* w_qkv  = (const float*)d_in[1];   // [2304,768]
    const float* w_proj = (const float*)d_in[2];   // [768,768]
    float* out = (float*)d_out;

    const long ROWS = (long)BATCH * NSEQ;          // 8192
    half_t* xh  = (half_t*)d_ws;                   // 8192*768
    half_t* wqh = xh  + ROWS * EMBED;              // 2304*768
    half_t* wph = wqh + (long)QKV_OUT * EMBED;     // 768*768
    half_t* qh  = wph + (long)EMBED * EMBED;       // [2,8,4096,96]
    half_t* kh  = qh  + ROWS * EMBED;
    half_t* vth = kh  + ROWS * EMBED;              // [2,8,96,4096] (V^T)
    half_t* ah  = vth + ROWS * EMBED;              // [2,4096,768]
    // total: 33,816,576 halfs = 64.5 MiB of d_ws

    const int CT = 256;
    cvt_f32_to_f16<<<(int)((ROWS * EMBED + CT - 1) / CT), CT, 0, stream>>>(x, xh, (int)(ROWS * EMBED));
    cvt_f32_to_f16<<<(QKV_OUT * EMBED + CT - 1) / CT, CT, 0, stream>>>(w_qkv, wqh, QKV_OUT * EMBED);
    cvt_f32_to_f16<<<(EMBED * EMBED + CT - 1) / CT, CT, 0, stream>>>(w_proj, wph, EMBED * EMBED);

    qkv_gemm_kernel<<<dim3(128, QKV_OUT / 64), 128, 0, stream>>>(xh, wqh, qh, kh, vth);
    flash_attn_kernel<<<dim3(BATCH * NHEAD, NSEQ / 128), 256, 0, stream>>>(qh, kh, vth, ah);
    proj_gemm_kernel<<<dim3(128, EMBED / 64), 128, 0, stream>>>(ah, wph, out);
}
